// GraphConvolution_72387378807298
// MI455X (gfx1250) — matrix operand order repaired and checked by reading, not compiled, not run on hardware
//
#include <hip/hip_runtime.h>
#include <stdint.h>

// ---------------------------------------------------------------------------
// GCNII-style layer for MI455X (gfx1250, wave32, WMMA)
//   out = theta * (concat(adj@x, x) @ W) + (1-theta) * ((1-a)x + a*h0 + 0.1*agg)
// Heavy GEMMs: v_wmma_f32_16x16x32_bf16, double-buffered BK=64 LDS slabs,
// tiles staged with global_load_async_to_lds_b128 (ASYNCcnt path).
// ---------------------------------------------------------------------------

typedef __attribute__((ext_vector_type(16))) __bf16 v16bf;
typedef __attribute__((ext_vector_type(8)))  __bf16 v8bf;
typedef __attribute__((ext_vector_type(4)))  __bf16 v4bf;
typedef __attribute__((ext_vector_type(8)))  float  v8f;

#define NN 8192   // nodes
#define DD 256    // feature dim
#define KK2 512   // 2*DD
#define LSTR 72   // LDS row stride (bf16): 144B = 36 dwords, conflict-free, 16B-aligned

static __device__ __forceinline__ v16bf combine16(v8bf lo, v8bf hi) {
  v16bf r;
#pragma unroll
  for (int i = 0; i < 8; ++i) { r[i] = lo[i]; r[i + 8] = hi[i]; }
  return r;
}

// Wave-relative LDS byte offset: flat shared address truncated to 32 bits
// (in the LDS aperture the low 32 address bits are the LDS offset).
static __device__ __forceinline__ uint32_t lds_off(const void* p) {
  return (uint32_t)(uintptr_t)p;
}

// CDNA5 async global->LDS copies (ASYNCcnt-tracked, no VGPR round trip).
static __device__ __forceinline__ void async_ld_b128(uint32_t lds, const void* g) {
  asm volatile("global_load_async_to_lds_b128 %0, %1, off"
               :: "v"(lds), "v"(g) : "memory");
}
static __device__ __forceinline__ void wait_async0() {
  asm volatile("s_wait_asynccnt 0x0" ::: "memory");
}

// ------------------------- prepass: f32 -> bf16 ----------------------------
__global__ __launch_bounds__(256)
void cvt_bf16_kernel(const float* __restrict__ s, __bf16* __restrict__ d, int n) {
  int i = (blockIdx.x * 256 + threadIdx.x) * 4;
  if (i + 3 < n) {
    float4 v = *(const float4*)(s + i);
    v4bf p;
    p[0] = (__bf16)v.x; p[1] = (__bf16)v.y; p[2] = (__bf16)v.z; p[3] = (__bf16)v.w;
    *(v4bf*)(d + i) = p;
  }
}

// -------------------- prepass: transpose + f32 -> bf16 ---------------------
// src: [R][C] row-major f32   ->   dst: [C][R] row-major bf16
__global__ __launch_bounds__(256)
void transpose_bf16_kernel(const float* __restrict__ src, __bf16* __restrict__ dst,
                           int R, int C) {
  __shared__ float tile[32][33];
  const int c0 = blockIdx.x * 32, r0 = blockIdx.y * 32;
  const int lc = threadIdx.x & 31, lr = threadIdx.x >> 5;  // 8 rows / pass
#pragma unroll
  for (int i = 0; i < 4; ++i)
    tile[lr + i * 8][lc] = src[(size_t)(r0 + lr + i * 8) * C + c0 + lc];
  __syncthreads();
#pragma unroll
  for (int i = 0; i < 4; ++i)
    dst[(size_t)(c0 + lr + i * 8) * R + r0 + lc] = (__bf16)tile[lc][lr + i * 8];
}

// --------------------- neighbor sampling (top-3 gumbel) --------------------
// Top-k of gumbel(-log(-log u)) == top-k of u (monotonic); u = hash-based.
static __device__ __forceinline__ uint32_t hashu(uint32_t s) {
  s ^= s >> 16; s *= 0x7feb352dU; s ^= s >> 15; s *= 0x846ca68bU; s ^= s >> 16;
  return s;
}

__global__ __launch_bounds__(256)
void sample_topk_kernel(const float* __restrict__ adj, int* __restrict__ smp) {
  __shared__ float ss[8][32][3];
  __shared__ int   si[8][32][3];
  __shared__ int   sc[8][32];
  const int lane = threadIdx.x & 31, w = threadIdx.x >> 5;
  const int row = blockIdx.x * 8 + w;
  float s0 = -1.f, s1 = -1.f, s2 = -1.f;
  int i0 = 0, i1 = 0, i2 = 0, cnt = 0;
  const float* r = adj + (size_t)row * NN;
  for (int c = lane; c < NN; c += 32) {          // coalesced 128B per step
    float a = r[c];
    if (a > 0.f) {
      ++cnt;
      uint32_t h = hashu((uint32_t)(row * NN + c) ^ 0x9e3779b9U);
      float u = (float)(h >> 8) * (1.f / 16777216.f);
      if (u > s0)      { s2 = s1; i2 = i1; s1 = s0; i1 = i0; s0 = u; i0 = c; }
      else if (u > s1) { s2 = s1; i2 = i1; s1 = u;  i1 = c; }
      else if (u > s2) { s2 = u;  i2 = c; }
    }
  }
  ss[w][lane][0] = s0; ss[w][lane][1] = s1; ss[w][lane][2] = s2;
  si[w][lane][0] = i0; si[w][lane][1] = i1; si[w][lane][2] = i2;
  sc[w][lane] = cnt;
  __syncthreads();
  if (lane == 0) {
    float t0 = -1.f, t1 = -1.f, t2 = -1.f;
    int j0 = 0, j1 = 0, j2 = 0, tot = 0;
    for (int l2 = 0; l2 < 32; ++l2) {
      tot += sc[w][l2];
#pragma unroll
      for (int e = 0; e < 3; ++e) {
        float u = ss[w][l2][e]; int c = si[w][l2][e];
        if (u > t0)      { t2 = t1; j2 = j1; t1 = t0; j1 = j0; t0 = u; j0 = c; }
        else if (u > t1) { t2 = t1; j2 = j1; t1 = u;  j1 = c; }
        else if (u > t2) { t2 = u;  j2 = c; }
      }
    }
    smp[row * 4 + 0] = j0; smp[row * 4 + 1] = j1; smp[row * 4 + 2] = j2;
    smp[row * 4 + 3] = (tot < 3) ? tot : 3;
  }
}

// -------------------- GEMM 1: hi(bf16) = adj(f32) @ x ----------------------
// Block tile 32x256 (full output width -> adj streamed from HBM exactly once,
// xT resident in the 192MB L2). 8 waves: (wm 0..1) x (wn 0..3), 16x64 each.
// Double-buffered BK=64 slabs: B via async global->LDS, A via VGPR f32->bf16.
// Per loop body per wave: 8 WMMAs, 1 barrier.
__global__ __launch_bounds__(256)
void gemm1_adj_x_kernel(const float* __restrict__ adj, const __bf16* __restrict__ xT,
                        __bf16* __restrict__ hi) {
  __shared__ __bf16 As[2][32][LSTR];
  __shared__ __bf16 Bs[2][256][LSTR];
  const int t = threadIdx.x;
  const int lane = t & 31, wid = t >> 5;
  const int wm = wid & 1, wn = wid >> 1;
  const int m0 = blockIdx.x * 32;

  v8f acc[4];
#pragma unroll
  for (int i = 0; i < 4; ++i) acc[i] = (v8f){0.f, 0.f, 0.f, 0.f, 0.f, 0.f, 0.f, 0.f};

  const int arow = t >> 3, ac4 = t & 7;   // A: two float4 (2x16B) per thread
  const int brow = t >> 2, bch = t & 3;   // B: 2x16B per thread, x4 row passes

  const float* aptr = adj + (size_t)(m0 + arow) * NN;

  // stage B slab [k0, k0+64) into buffer buf (8 async b128 per thread)
  auto stageB = [&](int buf, int k0) {
#pragma unroll
    for (int i = 0; i < 4; ++i) {
      const int n = brow + i * 64;
      const __bf16* g = xT + (size_t)n * NN + k0;
      async_ld_b128(lds_off(&Bs[buf][n][bch * 8]),       g + bch * 8);
      async_ld_b128(lds_off(&Bs[buf][n][(bch + 4) * 8]), g + (bch + 4) * 8);
    }
  };
  auto loadA = [&](int k0, float4& v0, float4& v1) {
    const float4* p = (const float4*)(aptr + k0);
    v0 = p[ac4]; v1 = p[ac4 + 8];
    __builtin_prefetch(aptr + k0 + 128, 0, 3);
  };
  auto storeA = [&](int buf, float4 v0, float4 v1) {
    v4bf p0, p1;
    p0[0] = (__bf16)v0.x; p0[1] = (__bf16)v0.y; p0[2] = (__bf16)v0.z; p0[3] = (__bf16)v0.w;
    p1[0] = (__bf16)v1.x; p1[1] = (__bf16)v1.y; p1[2] = (__bf16)v1.z; p1[3] = (__bf16)v1.w;
    *(v4bf*)&As[buf][arow][ac4 * 4]       = p0;
    *(v4bf*)&As[buf][arow][(ac4 + 8) * 4] = p1;
  };

  // ---- prologue: stage slab 0 into buffer 0 ----
  float4 va0, va1;
  loadA(0, va0, va1);
  stageB(0, 0);
  storeA(0, va0, va1);
  wait_async0();
  __syncthreads();

  for (int k0 = 0; k0 < NN; k0 += 64) {
    const int cur = (k0 >> 6) & 1, nxt = cur ^ 1;
    const bool more = (k0 + 64) < NN;
    if (more) {   // kick off next slab while we compute this one
      loadA(k0 + 64, va0, va1);
      stageB(nxt, k0 + 64);
    }

    const int ml  = wm * 16 + (lane & 15);
    const int kbA = (lane < 16) ? 0 : 8;    // A frag K base (16x32 bf16 layout)
    const int kbB = (lane < 16) ? 0 : 8;    // B frag K base (32x16 bf16 layout)
#pragma unroll
    for (int kh = 0; kh < 2; ++kh) {        // two WMMA K-steps per slab
      const int ko = kh * 32;
      v16bf a = combine16(*(const v8bf*)&As[cur][ml][ko + kbA],
                          *(const v8bf*)&As[cur][ml][ko + kbA + 16]);
      v16bf b[4];
#pragma unroll
      for (int ni = 0; ni < 4; ++ni) {
        const int n = wn * 64 + ni * 16 + (lane & 15);
        b[ni] = combine16(*(const v8bf*)&Bs[cur][n][ko + kbB],
                          *(const v8bf*)&Bs[cur][n][ko + kbB + 16]);
      }
#pragma unroll
      for (int ni = 0; ni < 4; ++ni)
        acc[ni] = __builtin_amdgcn_wmma_f32_16x16x32_bf16(
            false, a, false, b[ni], (short)0, acc[ni], false, false);
    }

    if (more) storeA(nxt, va0, va1);  // cvt after WMMAs hides global latency
    wait_async0();
    __syncthreads();
  }

  // C layout: VGPR r -> row r (lanes 0-15) / row 8+r (lanes 16-31), col = lane&15
  const int mrow = m0 + wm * 16 + ((lane < 16) ? 0 : 8);
  const int col0 = wn * 64 + (lane & 15);
#pragma unroll
  for (int ni = 0; ni < 4; ++ni)
#pragma unroll
    for (int r = 0; r < 8; ++r)
      hi[(size_t)(mrow + r) * DD + col0 + ni * 16] = (__bf16)acc[ni][r];
}

// ------- GEMM 2 + fused epilogue: out = th*(support@W) + (1-th)*r ----------
// support = [hi | x] (bf16). Both slabs staged fully asynchronously, BK=64.
__global__ __launch_bounds__(256)
void gemm2_out_kernel(const __bf16* __restrict__ hi, const __bf16* __restrict__ xbf,
                      const __bf16* __restrict__ wT, const float* __restrict__ x,
                      const float* __restrict__ h0, const int* __restrict__ smp,
                      const float* __restrict__ lam, const float* __restrict__ alp,
                      const int* __restrict__ lint, float* __restrict__ out) {
  __shared__ __bf16 As[2][32][LSTR];
  __shared__ __bf16 Bs[2][256][LSTR];
  const int t = threadIdx.x;
  const int lane = t & 31, wid = t >> 5;
  const int wm = wid & 1, wn = wid >> 1;
  const int m0 = blockIdx.x * 32;

  v8f acc[4];
#pragma unroll
  for (int i = 0; i < 4; ++i) acc[i] = (v8f){0.f, 0.f, 0.f, 0.f, 0.f, 0.f, 0.f, 0.f};

  const int arow = t >> 3, ach = t & 7;   // A: one 16B chunk per thread
  const int brow = t >> 2, bch = t & 3;   // B: 2x16B per thread, x4 row passes

  // stage slab [k0, k0+64) into buffer buf (concat boundary is BK-aligned)
  auto stage = [&](int buf, int k0) {
    const __bf16* srcA = (k0 < DD) ? hi : xbf;   // concat(hi, x)
    const int kc = k0 & (DD - 1);
    async_ld_b128(lds_off(&As[buf][arow][ach * 8]),
                  srcA + (size_t)(m0 + arow) * DD + kc + ach * 8);
#pragma unroll
    for (int i = 0; i < 4; ++i) {
      const int n = brow + i * 64;
      const __bf16* g = wT + (size_t)n * KK2 + k0;
      async_ld_b128(lds_off(&Bs[buf][n][bch * 8]),       g + bch * 8);
      async_ld_b128(lds_off(&Bs[buf][n][(bch + 4) * 8]), g + (bch + 4) * 8);
    }
  };

  stage(0, 0);
  wait_async0();
  __syncthreads();

  for (int k0 = 0; k0 < KK2; k0 += 64) {
    const int cur = (k0 >> 6) & 1, nxt = cur ^ 1;
    if (k0 + 64 < KK2) stage(nxt, k0 + 64);

    const int ml  = wm * 16 + (lane & 15);
    const int kbA = (lane < 16) ? 0 : 8;
    const int kbB = (lane < 16) ? 0 : 8;
#pragma unroll
    for (int kh = 0; kh < 2; ++kh) {
      const int ko = kh * 32;
      v16bf a = combine16(*(const v8bf*)&As[cur][ml][ko + kbA],
                          *(const v8bf*)&As[cur][ml][ko + kbA + 16]);
      v16bf b[4];
#pragma unroll
      for (int ni = 0; ni < 4; ++ni) {
        const int n = wn * 64 + ni * 16 + (lane & 15);
        b[ni] = combine16(*(const v8bf*)&Bs[cur][n][ko + kbB],
                          *(const v8bf*)&Bs[cur][n][ko + kbB + 16]);
      }
#pragma unroll
      for (int ni = 0; ni < 4; ++ni)
        acc[ni] = __builtin_amdgcn_wmma_f32_16x16x32_bf16(
            false, a, false, b[ni], (short)0, acc[ni], false, false);
    }

    wait_async0();
    __syncthreads();
  }

  const float lamv = lam[0], alv = alp[0];
  const float theta = fminf(1.f, logf(lamv / (float)lint[0] + 1.f));
  const int mrow = m0 + wm * 16 + ((lane < 16) ? 0 : 8);
  const int col0 = wn * 64 + (lane & 15);
#pragma unroll
  for (int ni = 0; ni < 4; ++ni) {
    const int n = col0 + ni * 16;
#pragma unroll
    for (int r = 0; r < 8; ++r) {
      const int m = mrow + r;
      const int* sp = smp + m * 4;
      const int cnt = sp[3];
      float loc = 0.f;
      if (cnt > 0) loc += x[(size_t)sp[0] * DD + n];
      if (cnt > 1) loc += x[(size_t)sp[1] * DD + n];
      if (cnt > 2) loc += x[(size_t)sp[2] * DD + n];
      loc *= (cnt > 0) ? (1.f / (float)cnt) : 0.f;
      const float rr = (1.f - alv) * x[(size_t)m * DD + n] +
                       alv * h0[(size_t)m * DD + n] + 0.1f * loc;
      out[(size_t)m * DD + n] = theta * acc[ni][r] + (1.f - theta) * rr;
    }
  }
}

// ---------------------------------------------------------------------------
extern "C" void kernel_launch(void* const* d_in, const int* in_sizes, int n_in,
                              void* d_out, int out_size, void* d_ws, size_t ws_size,
                              hipStream_t stream) {
  const float* x    = (const float*)d_in[0];
  const float* adj  = (const float*)d_in[1];
  const float* h0   = (const float*)d_in[2];
  const float* w    = (const float*)d_in[3];
  const float* lam  = (const float*)d_in[4];
  const float* alp  = (const float*)d_in[5];
  const int*   lint = (const int*)d_in[6];
  float* out = (float*)d_out;

  char* ws = (char*)d_ws;
  __bf16* x_bf  = (__bf16*)(ws);                                   // 4 MB
  __bf16* xT_bf = (__bf16*)(ws + (size_t)(4u << 20));              // 4 MB
  __bf16* hi_bf = (__bf16*)(ws + (size_t)(8u << 20));              // 4 MB
  __bf16* wT_bf = (__bf16*)(ws + (size_t)(12u << 20));             // 256 KB
  int*    smp   = (int*)   (ws + (size_t)(12u << 20) + KK2 * DD * 2);

  // prepasses
  cvt_bf16_kernel<<<(NN * DD) / (256 * 4), 256, 0, stream>>>(x, x_bf, NN * DD);
  transpose_bf16_kernel<<<dim3(DD / 32, NN / 32), 256, 0, stream>>>(x, xT_bf, NN, DD);
  transpose_bf16_kernel<<<dim3(DD / 32, KK2 / 32), 256, 0, stream>>>(w, wT_bf, KK2, DD);
  sample_topk_kernel<<<NN / 8, 256, 0, stream>>>(adj, smp);

  // hi = adj @ x   (bf16 WMMA; adj streamed once, double-buffered + async)
  gemm1_adj_x_kernel<<<NN / 32, 256, 0, stream>>>(adj, xT_bf, hi_bf);

  // out = theta*(concat(hi,x)@W) + (1-theta)*r   (bf16 WMMA + fused epilogue)
  gemm2_out_kernel<<<NN / 32, 256, 0, stream>>>(hi_bf, x_bf, wT_bf, x, h0, smp,
                                                lam, alp, lint, out);
}
